// _EmformerLayer_37950331027900
// MI455X (gfx1250) — matrix operand order repaired and checked by reading, not compiled, not run on hardware
//
#include <hip/hip_runtime.h>
#include <cstdint>

typedef __attribute__((ext_vector_type(16))) __bf16 v16bf;
typedef __attribute__((ext_vector_type(8)))  float  v8f;

// ---- problem constants ----
constexpr int D = 1024, H = 16, FFNI = 4096, SEG = 128, RCL = 4, TT = 2048, BB = 2;
constexpr int NN = TT / SEG;       // 16
constexpr int MM = NN - 1;         // 15
constexpr int RR = RCL * NN;       // 64
constexpr int SS = NN;             // 16
constexpr int QQ = RR + TT + SS;   // 2128
constexpr int KVV = MM + RR + TT;  // 2127
constexpr int DH = D / H;          // 64
constexpr float SCALING = 0.125f;  // dh^-0.5
constexpr float NEGINF = -1.0e8f;

constexpr int RTB = (RR + TT) * BB;  // 4224 rows
constexpr int QB  = QQ * BB;         // 4256 rows
constexpr int KVB = KVV * BB;        // 4254 rows
constexpr int QTILES = QQ / 16;      // 133 (exact)
constexpr int KVPAD  = ((KVV + 31) / 32) * 32;  // 2144

// ---- workspace layout (float offsets) ----
constexpr size_t OFF_MEMS = 0;
constexpr size_t OFF_LN   = OFF_MEMS + (size_t)MM * BB * D;
constexpr size_t OFF_SUM  = OFF_LN   + (size_t)(RR + TT) * BB * D;
constexpr size_t OFF_QRY  = OFF_SUM  + (size_t)NN * BB * D;
constexpr size_t OFF_KV   = OFF_QRY  + (size_t)QQ * BB * D;
constexpr size_t OFF_ATT  = OFF_KV   + (size_t)KVV * BB * 2 * D;
constexpr size_t OFF_OUT  = OFF_ATT  + (size_t)QQ * BB * D;
constexpr size_t OFF_RES  = OFF_OUT  + (size_t)QQ * BB * D;
constexpr size_t OFF_FLN  = OFF_RES  + (size_t)(RR + TT) * BB * D;
constexpr size_t OFF_H    = OFF_FLN  + (size_t)(RR + TT) * BB * D;
constexpr size_t OFF_FF   = OFF_H    + (size_t)(RR + TT) * BB * FFNI;

// ---- CDNA5 async global->LDS copy (ASYNCcnt path, CDNA5 ISA) ----
static __device__ __forceinline__ void async_ld16(void* lds_dst, const float* gsrc) {
    // flat shared pointer keeps the LDS byte offset in its low 32 bits
    uint32_t laddr = (uint32_t)(uintptr_t)lds_dst;
    uint64_t gaddr = (uint64_t)(uintptr_t)gsrc;
    asm volatile("global_load_async_to_lds_b128 %0, %1, off"
                 :: "v"(laddr), "v"(gaddr) : "memory");
}
// wait until <= n async ops outstanding (in-order completion => previous batch drained)
#define ASYNC_WAIT(n) asm volatile("s_wait_asynccnt " #n ::: "memory")

// ---- cross-lane XOR reduce via ds_swizzle (imm pattern, no index VGPR) ----
#define SWZ_XOR(x, m) \
    __int_as_float(__builtin_amdgcn_ds_swizzle(__float_as_int(x), (((m) << 10) | 0x1F)))

static __device__ __forceinline__ void fill8(v16bf& v, int base, float4 a, float4 b) {
    v[base + 0] = (__bf16)a.x; v[base + 1] = (__bf16)a.y;
    v[base + 2] = (__bf16)a.z; v[base + 3] = (__bf16)a.w;
    v[base + 4] = (__bf16)b.x; v[base + 5] = (__bf16)b.y;
    v[base + 6] = (__bf16)b.z; v[base + 7] = (__bf16)b.w;
}
static __device__ __forceinline__ float4 mul4(float4 a, float s) {
    return make_float4(a.x * s, a.y * s, a.z * s, a.w * s);
}

// ================= WMMA GEMM, 128x64 block tile, double-buffered async staging ==========
// C[M,N] = A[M,K] * W[N,K]^T + bias ; each wave: 16 rows x 64 cols (4 WMMA / K-step).
template <bool RELU>
__global__ __launch_bounds__(256) void gemm_wmma(const float* __restrict__ A,
                                                 const float* __restrict__ W,
                                                 const float* __restrict__ bias,
                                                 float* __restrict__ C,
                                                 int Mrows, int Ncols, int K) {
    __shared__ __align__(16) float a_lds[2][128][36];  // stride 36: conflict-free
    __shared__ __align__(16) float b_lds[2][64][36];

    const int t    = threadIdx.x;
    const int lane = t & 31;
    const int wv   = t >> 5;       // 8 waves
    const int hf   = lane >> 4;
    const int l16  = lane & 15;
    const int blockM = blockIdx.x * 128;
    const int blockN = blockIdx.y * 64;

    // fixed 6 async issues per thread per tile (clamped addresses keep count exact)
    auto stage = [&](int kb, int buf) {
#pragma unroll
        for (int j = 0; j < 4; j++) {                 // A: 128x32 = 1024 float4
            const int idx = j * 256 + t;
            const int row = idx >> 3;                 // 0..127
            const int c4  = (idx & 7) * 4;
            int grow = blockM + row;
            grow = grow < Mrows ? grow : (Mrows - 1);
            async_ld16(&a_lds[buf][row][c4], A + (size_t)grow * K + kb + c4);
        }
#pragma unroll
        for (int j = 0; j < 2; j++) {                 // W: 64x32 = 512 float4
            const int idx = j * 256 + t;
            const int row = idx >> 3;                 // 0..63
            const int c4  = (idx & 7) * 4;
            async_ld16(&b_lds[buf][row][c4], W + (size_t)(blockN + row) * K + kb + c4);
        }
    };

    v8f acc0 = {}, acc1 = {}, acc2 = {}, acc3 = {};
    stage(0, 0);
    for (int kb = 0; kb < K; kb += 32) {
        const int cur = (kb >> 5) & 1;
        if (kb + 32 < K) {
            stage(kb + 32, cur ^ 1);  // overlap next tile's copies with this tile's math
            ASYNC_WAIT(6);
        } else {
            ASYNC_WAIT(0);
        }
        __syncthreads();  // current buffer ready for all waves

        v16bf af;
        {
            const float* ar = &a_lds[cur][wv * 16 + l16][hf * 8];
            fill8(af, 0, ((const float4*)ar)[0], ((const float4*)ar)[1]);
            fill8(af, 8, ((const float4*)(ar + 16))[0], ((const float4*)(ar + 16))[1]);
        }
#pragma unroll
        for (int n = 0; n < 4; n++) {
            v16bf bf;
            const float* br = &b_lds[cur][n * 16 + l16][hf * 8];
            fill8(bf, 0, ((const float4*)br)[0], ((const float4*)br)[1]);
            fill8(bf, 8, ((const float4*)(br + 16))[0], ((const float4*)(br + 16))[1]);
            v8f& a = (n == 0 ? acc0 : n == 1 ? acc1 : n == 2 ? acc2 : acc3);
            a = __builtin_amdgcn_wmma_f32_16x16x32_bf16(false, af, false, bf,
                                                        (short)0, a, false, false);
        }
        __syncthreads();  // all waves done reading before buffer is overwritten
    }
#pragma unroll
    for (int n = 0; n < 4; n++) {
        const v8f& a = (n == 0 ? acc0 : n == 1 ? acc1 : n == 2 ? acc2 : acc3);
        const int ocol = blockN + n * 16 + l16;
#pragma unroll
        for (int r = 0; r < 8; r++) {
            const int orow = blockM + wv * 16 + r + hf * 8;
            if (orow < Mrows) {
                float v = a[r] + (bias ? bias[ocol] : 0.0f);
                if (RELU) v = fmaxf(v, 0.0f);
                C[(size_t)orow * Ncols + ocol] = v;
            }
        }
    }
}

// ================= Fused flash attention, double-buffered K/V staging =================
__global__ __launch_bounds__(256) void attn_kernel(const float* __restrict__ qbuf,
                                                   const float* __restrict__ kvbuf,
                                                   const unsigned char* __restrict__ amask,
                                                   const int* __restrict__ lengths,
                                                   float* __restrict__ ob) {
    __shared__ __align__(16) float klds[2][32][68];  // stride 68: conflict-free
    __shared__ __align__(16) float vlds[2][32][68];
    __shared__ __bf16 pshare[8][16 * 32];

    const int t    = threadIdx.x;
    const int lane = t & 31;
    const int wv   = t >> 5;
    const int hf   = lane >> 4;
    const int l16  = lane & 15;
    const int qt0 = blockIdx.x * 8 + wv;
    const bool qvalid = qt0 < QTILES;
    const int qt = qvalid ? qt0 : QTILES - 1;  // keep all waves alive for barriers
    const int bh = blockIdx.y;
    const int b = bh / H, hh = bh % H;
    const int dbase = hh * DH;

    int maxlen = lengths[0];
#pragma unroll
    for (int i = 1; i < BB; i++) maxlen = max(maxlen, lengths[i]);
    const int klen = lengths[b] + MM + RR + TT - maxlen;

    const int q0 = qt * 16;
    // --- scaled q fragments (A-operand, K=64 -> two K-steps) ---
    v16bf qf0, qf1;
    {
        const float* qr = qbuf + ((size_t)(q0 + l16) * BB + b) * D + dbase;
        const int ka = hf * 8;
        const float4* p0 = (const float4*)(qr + ka);
        fill8(qf0, 0, mul4(p0[0], SCALING), mul4(p0[1], SCALING));
        fill8(qf0, 8, mul4(p0[4], SCALING), mul4(p0[5], SCALING));
        const float4* p1 = (const float4*)(qr + 32 + ka);
        fill8(qf1, 0, mul4(p1[0], SCALING), mul4(p1[1], SCALING));
        fill8(qf1, 8, mul4(p1[4], SCALING), mul4(p1[5], SCALING));
    }

    // fixed 4 async issues per thread per tile (clamped addresses)
    auto stage_kv = [&](int kvb, int buf) {
#pragma unroll
        for (int j = 0; j < 2; j++) {
            const int idx = j * 256 + t;
            const int row = idx >> 4;        // 0..31
            const int c4  = (idx & 15) * 4;  // 0..60
            int kvr = kvb + row;
            kvr = kvr < KVV ? kvr : (KVV - 1);
            const float* gk = kvbuf + ((size_t)kvr * BB + b) * (2 * D) + dbase + c4;
            async_ld16(&klds[buf][row][c4], gk);
            async_ld16(&vlds[buf][row][c4], gk + D);
        }
    };

    v8f o0 = {}, o1 = {}, o2 = {}, o3 = {};
    float mrow[8], lrow[8];
#pragma unroll
    for (int r = 0; r < 8; r++) { mrow[r] = -3.0e38f; lrow[r] = 0.0f; }

    stage_kv(0, 0);
    for (int kvb = 0; kvb < KVPAD; kvb += 32) {
        const int cur = (kvb >> 5) & 1;
        if (kvb + 32 < KVPAD) {
            stage_kv(kvb + 32, cur ^ 1);
            ASYNC_WAIT(4);
        } else {
            ASYNC_WAIT(0);
        }
        __syncthreads();

        // --- S = q @ k^T (16 x 32 via two 16x16 WMMA C-tiles, K=64) ---
        float sreg[2][8];
#pragma unroll
        for (int nt = 0; nt < 2; nt++) {
            v8f sacc = {};
#pragma unroll
            for (int s = 0; s < 2; s++) {
                v16bf kf;
                const float* krow = &klds[cur][nt * 16 + l16][s * 32 + hf * 8];
                fill8(kf, 0, ((const float4*)krow)[0], ((const float4*)krow)[1]);
                fill8(kf, 8, ((const float4*)(krow + 16))[0], ((const float4*)(krow + 16))[1]);
                sacc = __builtin_amdgcn_wmma_f32_16x16x32_bf16(false, s == 0 ? qf0 : qf1,
                                                               false, kf, (short)0, sacc,
                                                               false, false);
            }
#pragma unroll
            for (int r = 0; r < 8; r++) {
                const int qi = q0 + r + hf * 8;
                const int col = kvb + nt * 16 + l16;
                bool blocked = (col >= KVV) || (col >= klen);
                if (!blocked) blocked = (amask[(size_t)qi * KVV + col] != 0);
                sreg[nt][r] = blocked ? NEGINF : sacc[r];
            }
        }
        // --- online softmax: 16-lane XOR reductions via ds_swizzle ---
#pragma unroll
        for (int r = 0; r < 8; r++) {
            float mx = fmaxf(sreg[0][r], sreg[1][r]);
            mx = fmaxf(mx, SWZ_XOR(mx, 1));
            mx = fmaxf(mx, SWZ_XOR(mx, 2));
            mx = fmaxf(mx, SWZ_XOR(mx, 4));
            mx = fmaxf(mx, SWZ_XOR(mx, 8));
            const float mn = fmaxf(mrow[r], mx);
            const float sc = __expf(mrow[r] - mn);
            float p0v = __expf(sreg[0][r] - mn);
            float p1v = __expf(sreg[1][r] - mn);
            sreg[0][r] = p0v;
            sreg[1][r] = p1v;
            float rs = p0v + p1v;
            rs += SWZ_XOR(rs, 1);
            rs += SWZ_XOR(rs, 2);
            rs += SWZ_XOR(rs, 4);
            rs += SWZ_XOR(rs, 8);
            lrow[r] = lrow[r] * sc + rs;
            mrow[r] = mn;
            o0[r] *= sc; o1[r] *= sc; o2[r] *= sc; o3[r] *= sc;
        }
        // --- transpose P (C-layout -> A-fragment) through per-wave LDS scratch ---
        __bf16* ps = &pshare[wv][0];
#pragma unroll
        for (int r = 0; r < 8; r++) {
            ps[(r + hf * 8) * 32 + l16]      = (__bf16)sreg[0][r];
            ps[(r + hf * 8) * 32 + 16 + l16] = (__bf16)sreg[1][r];
        }
        v16bf pf;
        {
            const int ka = hf * 8;
#pragma unroll
            for (int i = 0; i < 8; i++) pf[i] = ps[l16 * 32 + ka + i];
#pragma unroll
            for (int i = 0; i < 8; i++) pf[8 + i] = ps[l16 * 32 + 16 + ka + i];
        }
        // --- O += P @ V ---
#pragma unroll
        for (int nt2 = 0; nt2 < 4; nt2++) {
            v16bf vf;
#pragma unroll
            for (int i = 0; i < 16; i++)
                vf[i] = (__bf16)vlds[cur][(i < 8 ? hf * 8 + i : 8 + hf * 8 + i)][nt2 * 16 + l16];
            v8f& oo = (nt2 == 0 ? o0 : nt2 == 1 ? o1 : nt2 == 2 ? o2 : o3);
            oo = __builtin_amdgcn_wmma_f32_16x16x32_bf16(false, pf, false, vf, (short)0,
                                                         oo, false, false);
        }
        __syncthreads();  // done reading this buffer before next overwrite
    }
    if (qvalid) {  // wave-uniform predicate
#pragma unroll
        for (int r = 0; r < 8; r++) {
            const int orow = q0 + r + hf * 8;
            const float inv = 1.0f / lrow[r];
            float* dst = ob + ((size_t)orow * BB + b) * D + dbase + l16;
            dst[0]  = o0[r] * inv;
            dst[16] = o1[r] * inv;
            dst[32] = o2[r] * inv;
            dst[48] = o3[r] * inv;
        }
    }
}

// ================= LayerNorm kernels (one block = one (pos,b) row, D=1024) =================
__global__ __launch_bounds__(256) void ln_in_kernel(const float* __restrict__ rc,
                                                    const float* __restrict__ utt,
                                                    const float* __restrict__ g,
                                                    const float* __restrict__ be,
                                                    float* __restrict__ out) {
    __shared__ float r1[256], r2[256];
    const int row = blockIdx.x;  // 0..RTB-1
    const float* x = (row < RR * BB) ? (rc + (size_t)row * D)
                                     : (utt + (size_t)(row - RR * BB) * D);
    const int t = threadIdx.x;
    float s = 0.f, s2 = 0.f;
#pragma unroll
    for (int i = 0; i < 4; i++) {
        float v = x[t + i * 256];
        s += v; s2 += v * v;
    }
    r1[t] = s; r2[t] = s2; __syncthreads();
    for (int off = 128; off > 0; off >>= 1) {
        if (t < off) { r1[t] += r1[t + off]; r2[t] += r2[t + off]; }
        __syncthreads();
    }
    const float mean = r1[0] / D;
    const float var = r2[0] / D - mean * mean;
    const float rs = rsqrtf(var + 1e-5f);
#pragma unroll
    for (int i = 0; i < 4; i++) {
        int c = t + i * 256;
        out[(size_t)row * D + c] = (x[c] - mean) * rs * g[c] + be[c];
    }
}

__global__ __launch_bounds__(256) void ln_plain_kernel(const float* __restrict__ in,
                                                       const float* __restrict__ g,
                                                       const float* __restrict__ be,
                                                       float* __restrict__ out) {
    __shared__ float r1[256], r2[256];
    const int row = blockIdx.x;
    const float* x = in + (size_t)row * D;
    const int t = threadIdx.x;
    float s = 0.f, s2 = 0.f;
#pragma unroll
    for (int i = 0; i < 4; i++) {
        float v = x[t + i * 256];
        s += v; s2 += v * v;
    }
    r1[t] = s; r2[t] = s2; __syncthreads();
    for (int off = 128; off > 0; off >>= 1) {
        if (t < off) { r1[t] += r1[t + off]; r2[t] += r2[t + off]; }
        __syncthreads();
    }
    const float mean = r1[0] / D;
    const float var = r2[0] / D - mean * mean;
    const float rs = rsqrtf(var + 1e-5f);
#pragma unroll
    for (int i = 0; i < 4; i++) {
        int c = t + i * 256;
        out[(size_t)row * D + c] = (x[c] - mean) * rs * g[c] + be[c];
    }
}

__global__ __launch_bounds__(256) void final_ln_kernel(const float* __restrict__ res,
                                                       const float* __restrict__ ff,
                                                       const float* __restrict__ g,
                                                       const float* __restrict__ be,
                                                       float* __restrict__ dout) {
    __shared__ float xs[1024];
    __shared__ float r1[256], r2[256];
    const int row = blockIdx.x;
    const int t = threadIdx.x;
    float s = 0.f, s2 = 0.f;
#pragma unroll
    for (int i = 0; i < 4; i++) {
        int c = t + i * 256;
        float v = res[(size_t)row * D + c] + ff[(size_t)row * D + c];
        xs[c] = v; s += v; s2 += v * v;
    }
    r1[t] = s; r2[t] = s2; __syncthreads();
    for (int off = 128; off > 0; off >>= 1) {
        if (t < off) { r1[t] += r1[t + off]; r2[t] += r2[t + off]; }
        __syncthreads();
    }
    const float mean = r1[0] / D;
    const float var = r2[0] / D - mean * mean;
    const float rs = rsqrtf(var + 1e-5f);
    float* dst = (row < RR * BB)
                     ? (dout + (size_t)TT * BB * D + (size_t)row * D)
                     : (dout + (size_t)(row - RR * BB) * D);
#pragma unroll
    for (int i = 0; i < 4; i++) {
        int c = t + i * 256;
        dst[c] = (xs[c] - mean) * rs * g[c] + be[c];
    }
}

// ================= small elementwise kernels =================
__global__ __launch_bounds__(256) void summary_kernel(const float* __restrict__ ln,
                                                      float* __restrict__ sum) {
    const int idx = blockIdx.x * 256 + threadIdx.x;
    if (idx >= NN * BB * D) return;
    const int n = idx / (BB * D);
    const int rem = idx % (BB * D);
    float acc = 0.f;
    for (int s = 0; s < SEG; s++)
        acc += ln[(size_t)(RR + n * SEG + s) * BB * D + rem];
    sum[idx] = acc * (1.0f / SEG);
}

__global__ __launch_bounds__(256) void resid_kernel(const float* __restrict__ outp,
                                                    const float* __restrict__ rc,
                                                    const float* __restrict__ utt,
                                                    float* __restrict__ res,
                                                    float* __restrict__ omems) {
    const size_t idx = (size_t)blockIdx.x * 256 + threadIdx.x;  // over QB*D
    const int qrow = (int)(idx / D);
    if (qrow < RTB) {
        float add = (qrow < RR * BB) ? rc[idx] : utt[idx - (size_t)RR * BB * D];
        res[idx] = outp[idx] + add;
    } else {
        float v = outp[idx];
        v = fminf(fmaxf(v, -10.0f), 10.0f);
        omems[idx - (size_t)RTB * D] = v;
    }
}

// ================= host launcher =================
extern "C" void kernel_launch(void* const* d_in, const int* in_sizes, int n_in,
                              void* d_out, int out_size, void* d_ws, size_t ws_size,
                              hipStream_t stream) {
    const float* utt     = (const float*)d_in[0];
    const float* rc      = (const float*)d_in[1];
    const float* mems    = (const float*)d_in[2];
    const float* ln_in_g = (const float*)d_in[3];
    const float* ln_in_b = (const float*)d_in[4];
    const float* Wq      = (const float*)d_in[5];
    const float* bq      = (const float*)d_in[6];
    const float* Wkv     = (const float*)d_in[7];
    const float* bkv     = (const float*)d_in[8];
    const float* Wo      = (const float*)d_in[9];
    const float* bo      = (const float*)d_in[10];
    const float* ffg     = (const float*)d_in[11];
    const float* ffb     = (const float*)d_in[12];
    const float* W1      = (const float*)d_in[13];
    const float* b1      = (const float*)d_in[14];
    const float* W2      = (const float*)d_in[15];
    const float* b2      = (const float*)d_in[16];
    const float* log_    = (const float*)d_in[17];
    const float* lob     = (const float*)d_in[18];
    const int* lengths   = (const int*)d_in[19];
    const unsigned char* amask = (const unsigned char*)d_in[20];

    float* ws   = (float*)d_ws;
    float* wm   = ws + OFF_MEMS;
    float* wln  = ws + OFF_LN;
    float* wsum = ws + OFF_SUM;
    float* wqry = ws + OFF_QRY;
    float* wkv  = ws + OFF_KV;
    float* watt = ws + OFF_ATT;
    float* wout = ws + OFF_OUT;
    float* wres = ws + OFF_RES;
    float* wfln = ws + OFF_FLN;
    float* wh   = ws + OFF_H;
    float* wff  = ws + OFF_FF;
    float* dout = (float*)d_out;

    // mems into ws so [mems | ln] is the contiguous KV GEMM input
    hipMemcpyAsync(wm, mems, (size_t)MM * BB * D * sizeof(float),
                   hipMemcpyDeviceToDevice, stream);

    ln_in_kernel<<<RTB, 256, 0, stream>>>(rc, utt, ln_in_g, ln_in_b, wln);
    summary_kernel<<<(NN * BB * D + 255) / 256, 256, 0, stream>>>(wln, wsum);

    gemm_wmma<false><<<dim3((QB + 127) / 128, D / 64), 256, 0, stream>>>(
        wln, Wq, bq, wqry, QB, D, D);
    gemm_wmma<false><<<dim3((KVB + 127) / 128, (2 * D) / 64), 256, 0, stream>>>(
        wm, Wkv, bkv, wkv, KVB, 2 * D, D);

    attn_kernel<<<dim3((QTILES + 7) / 8, BB * H), 256, 0, stream>>>(
        wqry, wkv, amask, lengths, watt);

    gemm_wmma<false><<<dim3((QB + 127) / 128, D / 64), 256, 0, stream>>>(
        watt, Wo, bo, wout, QB, D, D);

    resid_kernel<<<(QB * D) / 256, 256, 0, stream>>>(
        wout, rc, utt, wres, dout + (size_t)(TT + RR) * BB * D);

    ln_plain_kernel<<<RTB, 256, 0, stream>>>(wres, ffg, ffb, wfln);

    gemm_wmma<true><<<dim3((RTB + 127) / 128, FFNI / 64), 256, 0, stream>>>(
        wfln, W1, b1, wh, RTB, FFNI, D);
    gemm_wmma<false><<<dim3((RTB + 127) / 128, D / 64), 256, 0, stream>>>(
        wh, W2, b2, wff, RTB, D, FFNI);

    final_ln_kernel<<<RTB, 256, 0, stream>>>(wres, wff, log_, lob, dout);
}
